// VoxelClustering_44014824849718
// MI455X (gfx1250) — compile-verified, matrix operand order checked by reading, not run on hardware
//
#include <hip/hip_runtime.h>
#include <cstdint>

// ---------------- problem constants (from reference) ----------------
static constexpr int     Bdim = 32, Gdim = 128, Pdim = 2048;
static constexpr int64_t N    = (int64_t)Bdim * Gdim * Pdim;   // 8388608
static constexpr int     HB   = 4096;          // radix blocks per pass
static constexpr int     HEPB = 2048;          // elements per radix block
static constexpr int     SCT  = 128;           // scatter threads (16 elems/thread)
static constexpr int     HIST_TOTAL = 256 * HB;          // 1048576 entries
static constexpr int     SCAN_BLKS  = HIST_TOTAL / 1024; // 1024 scan blocks
static constexpr int     UB   = 2048;          // unique-stage blocks
static constexpr int     UEPB = 4096;          // elements per unique block

typedef __attribute__((ext_vector_type(16))) _Float16 v16h;
typedef __attribute__((ext_vector_type(8)))  float    v8f;

// ---------------- CDNA5 async global->LDS helpers ----------------
__device__ __forceinline__ void async_load_b128(uint32_t lds_off, const void* gptr) {
  asm volatile("global_load_async_to_lds_b128 %0, %1, off"
               :: "v"(lds_off), "v"((uint64_t)(uintptr_t)gptr)
               : "memory");
}
__device__ __forceinline__ void wait_asynccnt0() {
  asm volatile("s_wait_asynccnt 0" ::: "memory");
}

// ---------------- 1) key build ----------------
__global__ void build_keys_kernel(const float* __restrict__ pred,
                                  const unsigned char* __restrict__ mask,
                                  uint64_t* __restrict__ keys,
                                  uint32_t* __restrict__ pay) {
  int64_t gid = (int64_t)blockIdx.x * blockDim.x + threadIdx.x;
  int64_t i0  = gid * 4;                     // 4 points / thread, 48B aligned
  const float4* p4 = (const float4*)(pred + i0 * 3);
  float4 A = p4[0], Bv = p4[1], C = p4[2];
  uchar4 m4 = *(const uchar4*)(mask + i0);
  float xs[4] = {A.x, A.w, Bv.z, C.y};
  float ys[4] = {A.y, Bv.x, Bv.w, C.z};
  float zs[4] = {A.z, Bv.y, C.x, C.w};
  unsigned char ms[4] = {m4.x, m4.y, m4.z, m4.w};
#pragma unroll
  for (int k = 0; k < 4; ++k) {
    int64_t i = i0 + k;
    uint64_t bg = (uint64_t)(i >> 11);       // i / P, P = 2048
    int vx = (int)floorf(xs[k] / 0.05f) + 32768;
    int vy = (int)floorf(ys[k] / 0.05f) + 32768;
    int vz = (int)floorf(zs[k] / 0.05f) + 32768;
    vx = min(max(vx, 0), 65535);
    vy = min(max(vy, 0), 65535);
    vz = min(max(vz, 0), 65535);
    uint64_t key = (bg << 48) | ((uint64_t)vx << 32) | ((uint64_t)vy << 16) | (uint64_t)vz;
    if (!ms[k]) key = 0x7FFFFFFFFFFFFFFFull; // sentinel (int64 max)
    keys[i] = key;
    pay[i]  = (uint32_t)i;
  }
}

// ---------------- 2a) per-pass histogram (async-to-LDS staged) ----------------
__global__ void hist_kernel(const uint64_t* __restrict__ keys,
                            uint32_t* __restrict__ hist, int shift) {
  __shared__ uint64_t tile[HEPB];            // 16 KB staged asynchronously
  __shared__ uint32_t lh[256];
  int t = threadIdx.x;
  lh[t] = 0;
  int64_t base = (int64_t)blockIdx.x * HEPB;
#pragma unroll
  for (int c = 0; c < 4; ++c) {              // 4 chunks * 2 keys * 256 thr = 2048 keys
    int e = (c * 256 + t) * 2;
    async_load_b128((uint32_t)(uintptr_t)&tile[e], &keys[base + e]);
  }
  wait_asynccnt0();
  __syncthreads();
#pragma unroll
  for (int j = 0; j < 8; ++j) {
    uint64_t k = tile[t * 8 + j];
    unsigned d = (unsigned)((k >> shift) & 0xFF);
    atomicAdd(&lh[d], 1u);                   // ds_add_u32 (wave32 LDS atomics)
  }
  __syncthreads();
  hist[(int64_t)t * HB + blockIdx.x] = lh[t];   // digit-major layout
}

// ---------------- 2b) parallel 2-level scan of hist ----------------
// scan1: 1024 blocks x 256 thr, 4 entries/thr -> in-place local exclusive scan
__global__ void scan1_kernel(uint32_t* __restrict__ hist,
                             uint32_t* __restrict__ partials) {
  __shared__ uint32_t s[256];
  int t = threadIdx.x;
  int64_t base = (int64_t)blockIdx.x * 1024 + t * 4;
  uint4 v = *(const uint4*)(hist + base);
  uint32_t sum = v.x + v.y + v.z + v.w;
  s[t] = sum;
  __syncthreads();
  for (int off = 1; off < 256; off <<= 1) {  // Hillis-Steele inclusive
    uint32_t u = (t >= off) ? s[t - off] : 0;
    __syncthreads();
    s[t] += u;
    __syncthreads();
  }
  uint32_t eb = (t == 0) ? 0 : s[t - 1];
  uint4 o;
  o.x = eb; o.y = eb + v.x; o.z = eb + v.x + v.y; o.w = eb + v.x + v.y + v.z;
  *(uint4*)(hist + base) = o;
  if (t == 255) partials[blockIdx.x] = s[255];
}

// scan2: single block over the 1024 block totals -> exclusive
__global__ void scan2_kernel(uint32_t* __restrict__ partials) {
  __shared__ uint32_t s[1024];
  int t = threadIdx.x;
  uint32_t v = partials[t];
  s[t] = v;
  __syncthreads();
  for (int off = 1; off < 1024; off <<= 1) {
    uint32_t u = (t >= off) ? s[t - off] : 0;
    __syncthreads();
    s[t] += u;
    __syncthreads();
  }
  partials[t] = (t == 0) ? 0 : s[t - 1];
}

// ---------------- 2c) stable scatter with WMMA ranking scan ----------------
// 128 threads, 16 consecutive elems each. cnt[t][d] <= 16 (exact in f16);
// per-digit 128-thread exclusive scan done as X(8x16) * UpperTri via
// v_wmma_f32_16x16x32_f16 (f32 accumulation exact: totals <= 2048 < 2^24).
// Keys staged once into LDS via async b128; global base = hist + partials.
__global__ void scatter_kernel(const uint64_t* __restrict__ keysIn,
                               const uint32_t* __restrict__ payIn,
                               uint64_t* __restrict__ keysOut,
                               uint32_t* __restrict__ payOut,
                               const uint32_t* __restrict__ hist,
                               const uint32_t* __restrict__ partials, int shift) {
  __shared__ uint16_t cnt[SCT][256];         // 64 KB
  __shared__ uint64_t tile[HEPB];            // 16 KB  (80 KB total, < 320 KB/WGP)
  int t    = threadIdx.x;                    // 0..127
  int lane = t & 31;
  int wave = t >> 5;                         // 4 waves
  int64_t gbase = (int64_t)blockIdx.x * HEPB;

  // stage this block's 2048 keys once (read global exactly once)
#pragma unroll
  for (int c = 0; c < 8; ++c) {              // 8 chunks * 2 keys * 128 thr
    int e = (c * 128 + t) * 2;
    async_load_b128((uint32_t)(uintptr_t)&tile[e], &keysIn[gbase + e]);
  }
  uint32_t* row32 = (uint32_t*)cnt[t];
  for (int k = 0; k < 128; ++k) row32[k] = 0;
  wait_asynccnt0();
  __syncthreads();

  // phase 1: sequential per-thread counting (keeps intra-thread stability)
  unsigned dpr[16];
#pragma unroll
  for (int j = 0; j < 16; ++j) {
    uint64_t k = tile[t * 16 + j];
    unsigned d = (unsigned)((k >> shift) & 0xFF);
    unsigned pr = cnt[t][d];
    cnt[t][d] = (uint16_t)(pr + 1);
    dpr[j] = d | (pr << 8);
  }
  __syncthreads();

  // B = upper-triangular ones (K<=N) in 16-bit B layout (32x16, K>=16 zero)
  int nlane = lane & 15, half = lane >> 4;
  v16h bU;
#pragma unroll
  for (int r = 0; r < 8; ++r) {
    int K0 = half * 8 + 2 * r;
    bU[2 * r]     = (r < 4 && K0     <= nlane) ? (_Float16)1 : (_Float16)0;
    bU[2 * r + 1] = (r < 4 && K0 + 1 <= nlane) ? (_Float16)1 : (_Float16)0;
  }
  int M = lane & 15;

  // phase 2: one WMMA per digit; wave w handles digits {it*4 + w}
  for (int it = 0; it < 64; ++it) {
    int d = it * 4 + wave;
    v16h a;
#pragma unroll
    for (int r = 0; r < 8; ++r) {
      if (r < 4 && M < 8) {
        int j0 = half * 8 + 2 * r;           // column (thread%16) index
        a[2 * r]     = (_Float16)(float)cnt[16 * M + j0][d];
        a[2 * r + 1] = (_Float16)(float)cnt[16 * M + j0 + 1][d];
      } else {
        a[2 * r] = (_Float16)0; a[2 * r + 1] = (_Float16)0;
      }
    }
    v8f c = {0.f, 0.f, 0.f, 0.f, 0.f, 0.f, 0.f, 0.f};
    v8f r8 = __builtin_amdgcn_wmma_f32_16x16x32_f16(
        false, a, false, bU, (short)0, c, false, false);

    // row totals: D[i][15] lives in lane 15, vgpr i (rows 0..7 used)
    float tot[8];
#pragma unroll
    for (int i = 0; i < 8; ++i) tot[i] = __shfl(r8[i], 15, 32);
    float P[8]; P[0] = 0.f;
#pragma unroll
    for (int i = 1; i < 8; ++i) P[i] = P[i - 1] + tot[i - 1];

    // write back exclusive prefix: lanes 0..15 hold rows 0..7, col = lane
    if (lane < 16) {
#pragma unroll
      for (int i = 0; i < 8; ++i) {
        int tt = 16 * i + lane;
        float X = (float)cnt[tt][d];
        cnt[tt][d] = (uint16_t)(P[i] + r8[i] - X);   // exact small ints
      }
    }
  }
  __syncthreads();

  // phase 3: scatter (global base = local-scanned hist + block partial)
#pragma unroll
  for (int j = 0; j < 16; ++j) {
    uint64_t k  = tile[t * 16 + j];
    unsigned d  = dpr[j] & 0xFF;
    unsigned pr = dpr[j] >> 8;
    uint32_t idx  = (uint32_t)d * HB + blockIdx.x;
    uint32_t dest = hist[idx] + partials[idx >> 10] + (uint32_t)cnt[t][d] + pr;
    keysOut[dest] = k;
    payOut[dest]  = payIn[gbase + t * 16 + j];
  }
}

// ---------------- 3) unique stage ----------------
__global__ void init_out_kernel(long long* __restrict__ outKeys) {
  int64_t i = (int64_t)blockIdx.x * blockDim.x + threadIdx.x;
  long long* row = outKeys + i * 5;          // decode(int64-max sentinel)
  row[0] = 255; row[1] = 127; row[2] = 32767; row[3] = 32767; row[4] = 32767;
}

__global__ void flag_count_kernel(const uint64_t* __restrict__ keys,
                                  uint32_t* __restrict__ bc) {
  __shared__ uint32_t s[256];
  int t = threadIdx.x;
  int64_t base = (int64_t)blockIdx.x * UEPB + (int64_t)t * 16;
  uint64_t prev = (base == 0) ? ~keys[0] : keys[base - 1];
  uint32_t c = 0;
#pragma unroll
  for (int j = 0; j < 16; ++j) {
    uint64_t k = keys[base + j];
    c += (k != prev);
    prev = k;
  }
  s[t] = c;
  __syncthreads();
  for (int off = 128; off > 0; off >>= 1) {
    if (t < off) s[t] += s[t + off];
    __syncthreads();
  }
  if (t == 0) bc[blockIdx.x] = s[0];
}

__global__ void scan_blocks_kernel(uint32_t* __restrict__ bc) {  // UB = 2048 entries
  __shared__ uint32_t sums[1024];
  int t = threadIdx.x;
  uint32_t a0 = bc[2 * t], a1 = bc[2 * t + 1];
  sums[t] = a0 + a1;
  __syncthreads();
  for (int off = 1; off < 1024; off <<= 1) {
    uint32_t v = (t >= off) ? sums[t - off] : 0;
    __syncthreads();
    sums[t] += v;
    __syncthreads();
  }
  uint32_t base = (t == 0) ? 0 : sums[t - 1];
  bc[2 * t]     = base;
  bc[2 * t + 1] = base + a0;
}

__global__ void scatter_unique_kernel(const uint64_t* __restrict__ keys,
                                      const uint32_t* __restrict__ pay,
                                      const uint32_t* __restrict__ bc,
                                      long long* __restrict__ outKeys,
                                      long long* __restrict__ outInv) {
  __shared__ uint32_t s[256];
  int t = threadIdx.x;
  int64_t base = (int64_t)blockIdx.x * UEPB + (int64_t)t * 16;
  uint64_t prev = (base == 0) ? ~keys[0] : keys[base - 1];
  uint32_t flags = 0, c = 0;
#pragma unroll
  for (int j = 0; j < 16; ++j) {
    uint64_t k = keys[base + j];
    uint32_t f = (k != prev) ? 1u : 0u;
    flags |= f << j;
    c += f;
    prev = k;
  }
  s[t] = c;
  __syncthreads();
  for (int off = 1; off < 256; off <<= 1) {
    uint32_t v = (t >= off) ? s[t - off] : 0;
    __syncthreads();
    s[t] += v;
    __syncthreads();
  }
  uint32_t run = bc[blockIdx.x] + ((t == 0) ? 0 : s[t - 1]);
#pragma unroll
  for (int j = 0; j < 16; ++j) {
    uint64_t k = keys[base + j];
    uint32_t f = (flags >> j) & 1u;
    run += f;
    long long id = (long long)run - 1;
    if (f) {
      long long* row = outKeys + id * 5;
      long long vz = (long long)(k & 0xFFFF) - 32768;
      long long vy = (long long)((k >> 16) & 0xFFFF) - 32768;
      long long vx = (long long)((k >> 32) & 0xFFFF) - 32768;
      long long bg = (long long)(k >> 48);
      row[0] = bg >> 7;       // b = bg / G
      row[1] = bg & 127;      // g = bg % G
      row[2] = vx; row[3] = vy; row[4] = vz;
    }
    outInv[pay[base + j]] = id;
  }
}

// ---------------- launcher ----------------
extern "C" void kernel_launch(void* const* d_in, const int* in_sizes, int n_in,
                              void* d_out, int out_size, void* d_ws, size_t ws_size,
                              hipStream_t stream) {
  (void)in_sizes; (void)n_in; (void)out_size; (void)ws_size;
  const float*         pred = (const float*)d_in[0];
  const unsigned char* mask = (const unsigned char*)d_in[1];

  char* w = (char*)d_ws;
  uint64_t* keysA    = (uint64_t*)w; w += (size_t)N * 8;
  uint64_t* keysB    = (uint64_t*)w; w += (size_t)N * 8;
  uint32_t* payA     = (uint32_t*)w; w += (size_t)N * 4;
  uint32_t* payB     = (uint32_t*)w; w += (size_t)N * 4;
  uint32_t* hist     = (uint32_t*)w; w += (size_t)HIST_TOTAL * 4;
  uint32_t* partials = (uint32_t*)w; w += (size_t)SCAN_BLKS * 4;
  uint32_t* bc       = (uint32_t*)w; w += (size_t)UB * 4;

  long long* outKeys = (long long*)d_out;
  long long* outInv  = outKeys + (size_t)N * 5;

  build_keys_kernel<<<(int)(N / 1024), 256, 0, stream>>>(pred, mask, keysA, payA);

  uint64_t* ki = keysA; uint64_t* ko = keysB;
  uint32_t* pi = payA;  uint32_t* po = payB;
  for (int pass = 0; pass < 8; ++pass) {
    int shift = pass * 8;
    hist_kernel <<<HB, 256, 0, stream>>>(ki, hist, shift);
    scan1_kernel<<<SCAN_BLKS, 256, 0, stream>>>(hist, partials);
    scan2_kernel<<<1, 1024, 0, stream>>>(partials);
    scatter_kernel<<<HB, SCT, 0, stream>>>(ki, pi, ko, po, hist, partials, shift);
    uint64_t* tk = ki; ki = ko; ko = tk;
    uint32_t* tp = pi; pi = po; po = tp;
  }
  // 8 passes -> sorted data back in keysA / payA (ki / pi)

  init_out_kernel      <<<(int)(N / 256), 256, 0, stream>>>(outKeys);
  flag_count_kernel    <<<UB, 256, 0, stream>>>(ki, bc);
  scan_blocks_kernel   <<<1, 1024, 0, stream>>>(bc);
  scatter_unique_kernel<<<UB, 256, 0, stream>>>(ki, pi, bc, outKeys, outInv);
}
